// TopKSAE_61735859912747
// MI455X (gfx1250) — matrix operand order repaired and checked by reading, not compiled, not run on hardware
//
#include <hip/hip_runtime.h>

#define DIN   768
#define DSAE  12288
#define BATCH 8192
#define TOPK  64

typedef __attribute__((ext_vector_type(16))) __bf16 v16bf;
typedef __attribute__((ext_vector_type(8)))  float  v8f;

union FragAB { v16bf bf; unsigned int u[8]; };

__device__ __forceinline__ unsigned int pack_bf16x2(float lo, float hi) {
  // round-to-nearest-even f32 -> bf16, packed pair
  unsigned int a = __float_as_uint(lo);
  unsigned int b = __float_as_uint(hi);
  a = (a + 0x7FFFu + ((a >> 16) & 1u)) >> 16;
  b = (b + 0x7FFFu + ((b >> 16) & 1u)) >> 16;
  return (a & 0xFFFFu) | (b << 16);
}

__device__ __forceinline__ unsigned short f32_to_bf16(float f) {
  unsigned int u = __float_as_uint(f);
  u = (u + 0x7FFFu + ((u >> 16) & 1u)) >> 16;
  return (unsigned short)u;
}

// low 32 bits of a generic pointer to __shared__ = LDS byte offset
__device__ __forceinline__ unsigned lds_off(const void* p) {
  return (unsigned)(unsigned long long)p;
}

// ---------------------------------------------------------------------------
// Prep 1: A_bf16[row][k] = bf16(x - b_dec), packed pairs (one uint = 2 K).
// ---------------------------------------------------------------------------
__global__ __launch_bounds__(256) void sae_prep_a(
    const float* __restrict__ x, const float* __restrict__ bdec,
    unsigned int* __restrict__ Abf) {
  const int row = blockIdx.x;
  const float* xr = x + (size_t)row * DIN;
  unsigned int* ar = Abf + (size_t)row * (DIN / 2);
  for (int kp = threadIdx.x; kp < DIN / 2; kp += 256) {
    float v0 = xr[2 * kp + 0] - bdec[2 * kp + 0];
    float v1 = xr[2 * kp + 1] - bdec[2 * kp + 1];
    ar[kp] = pack_bf16x2(v0, v1);
  }
}

// ---------------------------------------------------------------------------
// Prep 2: Btb[n][k] = bf16(W_enc[k][n])  (transpose via 64x64 LDS tile)
// ---------------------------------------------------------------------------
__global__ __launch_bounds__(256) void sae_prep_bt(
    const float* __restrict__ Wenc, unsigned short* __restrict__ Btb) {
  __shared__ unsigned short tile[64][65];
  const int bn = blockIdx.x * 64;   // n tile base
  const int bk = blockIdx.y * 64;   // k tile base
  const int tid = threadIdx.x;
#pragma unroll
  for (int i = 0; i < 16; ++i) {
    int e = tid + i * 256;          // 0..4095
    int r = e >> 6;                 // k within tile
    int c = e & 63;                 // n within tile (coalesced)
    tile[r][c] = f32_to_bf16(Wenc[(size_t)(bk + r) * DSAE + bn + c]);
  }
  __syncthreads();
#pragma unroll
  for (int i = 0; i < 16; ++i) {
    int e = tid + i * 256;
    int nr = e >> 6;                // n within tile
    int kc = e & 63;                // k within tile (coalesced writes)
    Btb[(size_t)(bn + nr) * DIN + bk + kc] = tile[kc][nr];
  }
}

// ---------------------------------------------------------------------------
// Kernel: pre = relu(A_bf16 @ Btb^T + b_enc)   (i.e. (x-b_dec)@W_enc + b_enc)
// 64x64 tile / 128 threads (4 waves), BK=64 (2 WMMA K-steps per stage),
// double-buffered LDS fed by global_load_async_to_lds_b128 (ASYNCcnt).
// LDS rows padded to 36 dwords: 16B-aligned async writes, conflict-free
// fragment reads (r*36 mod 64 distinct for r in [0,16)).
// ---------------------------------------------------------------------------
#define ROWU 36                       // dwords per padded LDS row (64 bf16 + pad)
__global__ __launch_bounds__(128) void sae_encode_gemm(
    const unsigned short* __restrict__ Abf,   // [8192][768] bf16, centered
    const unsigned short* __restrict__ Btb,   // [12288][768] bf16, W_enc^T
    const float* __restrict__ benc,
    float* __restrict__ pre) {
  __shared__ unsigned int lds_a[2][64 * ROWU];
  __shared__ unsigned int lds_b[2][64 * ROWU];

  const int tid  = threadIdx.x;
  const int lane = tid & 31;
  const int wave = tid >> 5;
  const int wm = wave >> 1, wn = wave & 1;
  const int half = lane >> 4;            // K-split lane half
  const int l16  = lane & 15;
  const int bn = blockIdx.x * 64;
  const int bm = blockIdx.y * 64;

  const unsigned a0 = lds_off(&lds_a[0][0]), a1 = lds_off(&lds_a[1][0]);
  const unsigned b0 = lds_off(&lds_b[0][0]), b1 = lds_off(&lds_b[1][0]);

  v8f acc[2][2];
#pragma unroll
  for (int i = 0; i < 2; ++i)
#pragma unroll
    for (int j = 0; j < 2; ++j)
      acc[i][j] = (v8f){0.f, 0.f, 0.f, 0.f, 0.f, 0.f, 0.f, 0.f};

#define STAGE(buf, kt)                                                        \
  do {                                                                        \
    unsigned abase = (buf) ? a1 : a0, bbase = (buf) ? b1 : b0;                \
    _Pragma("unroll")                                                         \
    for (int t = 0; t < 4; ++t) {                                             \
      int ch = tid + t * 128;            /* 512 x 16B chunks */               \
      int row = ch >> 3, seg = ch & 7;                                        \
      unsigned ga = (unsigned)(((bm + row) * DIN + (kt) + seg * 8) * 2);      \
      unsigned la = abase + (unsigned)(row * (ROWU * 4) + seg * 16);          \
      asm volatile("global_load_async_to_lds_b128 %0, %1, %2 offset:0"        \
                   :: "v"(la), "v"(ga), "s"(Abf) : "memory");                 \
      unsigned gb = (unsigned)(((bn + row) * DIN + (kt) + seg * 8) * 2);      \
      unsigned lb = bbase + (unsigned)(row * (ROWU * 4) + seg * 16);          \
      asm volatile("global_load_async_to_lds_b128 %0, %1, %2 offset:0"        \
                   :: "v"(lb), "v"(gb), "s"(Btb) : "memory");                 \
    }                                                                         \
  } while (0)

  STAGE(0, 0);
  asm volatile("s_wait_asynccnt 0x0" ::: "memory");
  __syncthreads();

  int buf = 0;
  for (int kt = 0; kt < DIN; kt += 64, buf ^= 1) {
    if (kt + 64 < DIN) STAGE(buf ^ 1, kt + 64);

    const unsigned int* pa = &lds_a[buf][0];
    const unsigned int* pb = &lds_b[buf][0];
#pragma unroll
    for (int s = 0; s < 2; ++s) {          // two WMMA K-steps per stage
      FragAB afr[2], bfr[2];
#pragma unroll
      for (int mi = 0; mi < 2; ++mi) {
        int base = (wm * 32 + mi * 16 + l16) * ROWU + s * 16 + half * 8;
#pragma unroll
        for (int j = 0; j < 4; ++j) {
          afr[mi].u[j]     = pa[base + j];
          afr[mi].u[4 + j] = pa[base + 4 + j];
        }
      }
#pragma unroll
      for (int ni = 0; ni < 2; ++ni) {
        int base = (wn * 32 + ni * 16 + l16) * ROWU + s * 16 + half * 8;
#pragma unroll
        for (int j = 0; j < 8; ++j)
          bfr[ni].u[j] = pb[base + j];
      }
#pragma unroll
      for (int mi = 0; mi < 2; ++mi)
#pragma unroll
        for (int ni = 0; ni < 2; ++ni)
          acc[mi][ni] = __builtin_amdgcn_wmma_f32_16x16x32_bf16(
              false, afr[mi].bf, false, bfr[ni].bf, (short)0, acc[mi][ni],
              false, false);
    }

    asm volatile("s_wait_asynccnt 0x0" ::: "memory");
    __syncthreads();
  }
#undef STAGE

  // epilogue: +b_enc, ReLU, store (C layout per ISA 7.12.2, wave32)
#pragma unroll
  for (int mi = 0; mi < 2; ++mi) {
#pragma unroll
    for (int ni = 0; ni < 2; ++ni) {
      int col = bn + wn * 32 + ni * 16 + l16;
      float bias = benc[col];
#pragma unroll
      for (int v = 0; v < 8; ++v) {
        int row = bm + wm * 32 + mi * 16 + v + half * 8;
        float val = fmaxf(acc[mi][ni][v] + bias, 0.f);
        pre[(size_t)row * DSAE + col] = val;
      }
    }
  }
}

// ---------------------------------------------------------------------------
// Top-64 per row: exact 4-pass radix select on float bits (values >= 0 after
// ReLU so uint order == float order). Zeroes non-kept entries in place and
// emits compact (idx,val) lists into workspace.
// ---------------------------------------------------------------------------
__global__ __launch_bounds__(256) void sae_topk(
    float* __restrict__ sparse, int* __restrict__ ws_idx,
    float* __restrict__ ws_val) {
  __shared__ unsigned int rowbuf[DSAE];   // 48 KB row cache
  __shared__ unsigned int hist[256];
  __shared__ unsigned int sh_prefix, sh_need, sh_cnt, sh_pos;

  const int row = blockIdx.x;
  const int tid = threadIdx.x;
  float* rp = sparse + (size_t)row * DSAE;

  for (int i = tid; i < DSAE; i += 256)
    rowbuf[i] = __float_as_uint(rp[i]);
  if (tid < TOPK) { ws_idx[row * TOPK + tid] = 0; ws_val[row * TOPK + tid] = 0.f; }
  if (tid == 0) { sh_prefix = 0; sh_need = TOPK; sh_cnt = 0; sh_pos = 0; }
  __syncthreads();

  for (int shift = 24; shift >= 0; shift -= 8) {
    hist[tid] = 0;
    __syncthreads();
    unsigned int prefix = sh_prefix;
    unsigned int mask = (shift == 24) ? 0u : (0xFFFFFFFFu << (shift + 8));
    for (int i = tid; i < DSAE; i += 256) {
      unsigned int bb = rowbuf[i];
      if ((bb & mask) == prefix)
        atomicAdd(&hist[(bb >> shift) & 255u], 1u);
    }
    __syncthreads();
    if (tid == 0) {
      unsigned int need = sh_need;
      int d = 255;
      for (; d > 0; --d) {
        unsigned int c = hist[d];
        if (c >= need) break;
        need -= c;
      }
      sh_prefix = prefix | ((unsigned int)d << shift);
      sh_need = need;
    }
    __syncthreads();
  }

  const unsigned int T = sh_prefix;       // 64th-largest value bits
  const unsigned int need_eq = sh_need;   // ties at T to keep
  for (int i = tid; i < DSAE; i += 256) {
    unsigned int bb = rowbuf[i];
    bool keep = false;
    if (bb > T)       keep = true;
    else if (bb == T) keep = (atomicAdd(&sh_cnt, 1u) < need_eq);
    float v = keep ? __uint_as_float(bb) : 0.f;
    rp[i] = v;
    if (keep) {
      unsigned int pos = atomicAdd(&sh_pos, 1u);
      ws_idx[row * TOPK + pos] = i;
      ws_val[row * TOPK + pos] = v;
    }
  }
}

// ---------------------------------------------------------------------------
// Sparse decode: recon[row] = sum_k val_k * W_dec[idx_k,:] + b_dec
// (0.8 GFLOP total; W_dec is L2-resident at 37.7 MB)
// ---------------------------------------------------------------------------
__global__ __launch_bounds__(256) void sae_decode(
    const float* __restrict__ Wdec, const float* __restrict__ bdec,
    const int* __restrict__ ws_idx, const float* __restrict__ ws_val,
    float* __restrict__ recon) {
  __shared__ int   s_idx[TOPK];
  __shared__ float s_val[TOPK];
  const int row = blockIdx.x;
  const int tid = threadIdx.x;
  if (tid < TOPK) {
    s_idx[tid] = ws_idx[row * TOPK + tid];
    s_val[tid] = ws_val[row * TOPK + tid];
  }
  __syncthreads();
  const int c0 = tid, c1 = tid + 256, c2 = tid + 512;   // 3*256 = 768 cols
  float a0 = bdec[c0], a1 = bdec[c1], a2 = bdec[c2];
#pragma unroll 4
  for (int k = 0; k < TOPK; ++k) {
    float v = s_val[k];                  // wave-uniform
    if (v != 0.f) {
      const float* wr = Wdec + (size_t)s_idx[k] * DIN;
      a0 = fmaf(v, wr[c0], a0);
      a1 = fmaf(v, wr[c1], a1);
      a2 = fmaf(v, wr[c2], a2);
    }
  }
  float* op = recon + (size_t)row * DIN;
  op[c0] = a0; op[c1] = a1; op[c2] = a2;
}

// ---------------------------------------------------------------------------
extern "C" void kernel_launch(void* const* d_in, const int* in_sizes, int n_in,
                              void* d_out, int out_size, void* d_ws, size_t ws_size,
                              hipStream_t stream) {
  (void)in_sizes; (void)n_in; (void)out_size; (void)ws_size;
  const float* x    = (const float*)d_in[0];
  const float* Wenc = (const float*)d_in[1];
  const float* benc = (const float*)d_in[2];
  const float* Wdec = (const float*)d_in[3];
  const float* bdec = (const float*)d_in[4];

  float* recon  = (float*)d_out;                       // [8192, 768]
  float* sparse = recon + (size_t)BATCH * DIN;         // [8192, 12288]

  // workspace layout (~35.5 MB total)
  int*          ws_idx = (int*)d_ws;                                  // 2 MB
  float*        ws_val = (float*)(ws_idx + (size_t)BATCH * TOPK);     // 2 MB
  unsigned int* Abf    = (unsigned int*)(ws_val + (size_t)BATCH * TOPK);      // 12.6 MB
  unsigned short* Btb  = (unsigned short*)(Abf + (size_t)BATCH * DIN / 2);    // 18.9 MB

  sae_prep_a<<<BATCH, 256, 0, stream>>>(x, bdec, Abf);
  sae_prep_bt<<<dim3(DSAE / 64, DIN / 64), 256, 0, stream>>>(Wenc, Btb);
  sae_encode_gemm<<<dim3(DSAE / 64, BATCH / 64), 128, 0, stream>>>(
      (const unsigned short*)Abf, Btb, benc, sparse);
  sae_topk<<<BATCH, 256, 0, stream>>>(sparse, ws_idx, ws_val);
  sae_decode<<<BATCH, 256, 0, stream>>>(Wdec, bdec, ws_idx, ws_val, recon);
}
